// ContextAttentionalDecoder_33655363731501
// MI455X (gfx1250) — matrix operand order repaired and checked by reading, not compiled, not run on hardware
//
#include <hip/hip_runtime.h>
#include <hip/hip_bf16.h>
#include <stdint.h>

typedef __bf16 bhalf;
typedef __attribute__((ext_vector_type(16))) __bf16 v16bf;
typedef __attribute__((ext_vector_type(8)))  float  v8f;

#define BQ 32      // batch
#define TT 256     // time steps
#define SS 512     // encoder length
#define EE 512     // embed = hidden
#define NH 8       // heads
#define HD 64      // head dim
#define G4 2048    // 4*H
#define KC 2048    // combined K for gates (1024 ctx + 512 emb + 512 h)

// ---------------- WMMA tile loaders (CDNA5 16x16x32 bf16 layouts) ----------------

// A-matrix 16x32 (MxK): lanes 0-15 -> K={8*hi+0..7, 16+8*hi+0..7}; hi = lane>>4
__device__ __forceinline__ v16bf load_tileA(const bhalf* base, int stride, int lane) {
  int m = lane & 15, hi = lane >> 4;
  const bhalf* r = base + (size_t)m * stride + hi * 8;
  union { v16bf v; uint4 q[2]; } x;
  x.q[0] = *(const uint4*)(r);
  x.q[1] = *(const uint4*)(r + 16);
  return x.v;
}

// B-matrix 32x16 (KxN) from row-major W[N,K] (B[k,n]=W[n,k]): lane=n, K={8*hi+0..7, 16+8*hi+0..7}
__device__ __forceinline__ v16bf load_tileB(const bhalf* base, int stride, int lane) {
  int n = lane & 15, hi = lane >> 4;
  const bhalf* r = base + (size_t)n * stride + hi * 8;
  union { v16bf v; uint4 q[2]; } x;
  x.q[0] = *(const uint4*)(r);
  x.q[1] = *(const uint4*)(r + 16);
  return x.v;
}

union U8 { uint4 u; bhalf h[8]; };

// ---------------- one-time prep kernels ----------------

__global__ void build_wcomb_kernel(const float* __restrict__ W_ih,
                                   const float* __restrict__ W_hh,
                                   bhalf* __restrict__ Wc) {
  for (size_t i = (size_t)blockIdx.x * blockDim.x + threadIdx.x; i < (size_t)G4 * KC;
       i += (size_t)gridDim.x * blockDim.x) {
    int g = (int)(i >> 11), k = (int)(i & 2047);
    float v = (k < 1536) ? W_ih[(size_t)g * 1536 + k] : W_hh[(size_t)g * 512 + (k - 1536)];
    Wc[i] = (bhalf)v;
  }
}

__global__ void build_pair_kernel(const float* __restrict__ Wa, const float* __restrict__ Wb,
                                  bhalf* __restrict__ Wc, int n_per) {
  for (size_t i = (size_t)blockIdx.x * blockDim.x + threadIdx.x; i < (size_t)2 * n_per;
       i += (size_t)gridDim.x * blockDim.x)
    Wc[i] = (bhalf)((i < (size_t)n_per) ? Wa[i] : Wb[i - n_per]);
}

__global__ void cvt_kernel(const float* __restrict__ s, bhalf* __restrict__ d, size_t n) {
  for (size_t i = (size_t)blockIdx.x * blockDim.x + threadIdx.x; i < n;
       i += (size_t)gridDim.x * blockDim.x)
    d[i] = (bhalf)s[i];
}

__global__ void init_kernel(const float* __restrict__ h0, const float* __restrict__ c0,
                            float* __restrict__ h_f, float* __restrict__ c_f,
                            bhalf* __restrict__ h_bf) {
  int i = blockIdx.x * blockDim.x + threadIdx.x;
  if (i < BQ * EE) { h_f[i] = h0[i]; c_f[i] = c0[i]; h_bf[i] = (bhalf)h0[i]; }
}

__global__ void zero_kernel(int* __restrict__ p, int n) {
  int i = blockIdx.x * blockDim.x + threadIdx.x;
  if (i < n) p[i] = 0;
}

__global__ void tail_kernel(const float* __restrict__ h_f, const float* __restrict__ c_f,
                            float* __restrict__ out) {
  size_t base = (size_t)BQ * TT * 1536;
  int i = blockIdx.x * blockDim.x + threadIdx.x;
  if (i < BQ * EE) { out[base + i] = h_f[i]; out[base + BQ * EE + i] = c_f[i]; }
}

// ---------------- KV cache precompute: out[r,n] = sum_k enc[r,k]*W[n,k] + b[n] ----------------

__global__ __launch_bounds__(128) void kv_kernel(const bhalf* __restrict__ enc,
                                                 const bhalf* __restrict__ W,
                                                 const float* __restrict__ bias,
                                                 bhalf* __restrict__ out) {
  int lane = threadIdx.x & 31, wave = threadIdx.x >> 5;
  int id = blockIdx.x * 4 + wave;          // 32768 tile-waves: 1024 mtiles x 32 ntiles
  int mtile = id >> 5, ntile = id & 31;
  const bhalf* A0 = enc + (size_t)mtile * 16 * EE;
  const bhalf* B0 = W + (size_t)ntile * 16 * EE;
  v8f acc = {0.f, 0.f, 0.f, 0.f, 0.f, 0.f, 0.f, 0.f};
  for (int k0 = 0; k0 < EE; k0 += 32) {
    v16bf a = load_tileA(A0 + k0, EE, lane);
    v16bf b = load_tileB(B0 + k0, EE, lane);
    acc = __builtin_amdgcn_wmma_f32_16x16x32_bf16(false, a, false, b, (short)0, acc,
                                                  false, false);
  }
  int n = lane & 15, hi = lane >> 4, col = ntile * 16 + n;
  float bv = bias[col];
#pragma unroll
  for (int v = 0; v < 8; ++v)
    out[(size_t)(mtile * 16 + v + 8 * hi) * EE + col] = (bhalf)(acc[v] + bv);
}

// Same GEMM, but stores transposed for the V caches: VT[((b*NH+head)*HD+d)*SS + s]
__global__ __launch_bounds__(128) void kv_kernel_T(const bhalf* __restrict__ enc,
                                                   const bhalf* __restrict__ W,
                                                   const float* __restrict__ bias,
                                                   bhalf* __restrict__ outT) {
  int lane = threadIdx.x & 31, wave = threadIdx.x >> 5;
  int id = blockIdx.x * 4 + wave;
  int mtile = id >> 5, ntile = id & 31;
  const bhalf* A0 = enc + (size_t)mtile * 16 * EE;
  const bhalf* B0 = W + (size_t)ntile * 16 * EE;
  v8f acc = {0.f, 0.f, 0.f, 0.f, 0.f, 0.f, 0.f, 0.f};
  for (int k0 = 0; k0 < EE; k0 += 32) {
    v16bf a = load_tileA(A0 + k0, EE, lane);
    v16bf b = load_tileB(B0 + k0, EE, lane);
    acc = __builtin_amdgcn_wmma_f32_16x16x32_bf16(false, a, false, b, (short)0, acc,
                                                  false, false);
  }
  int n = lane & 15, hi = lane >> 4, col = ntile * 16 + n;  // col = head*64 + d
  float bv = bias[col];
#pragma unroll
  for (int v = 0; v < 8; ++v) {
    int row = mtile * 16 + v + 8 * hi;     // row = b*512 + s
    int b = row >> 9, s = row & 511;
    outT[((size_t)(b * NH) * HD + col) * SS + s] = (bhalf)(acc[v] + bv);
  }
}

// ---------------- per-step: fused gates GEMM + LSTM cell + Q projection ----------------
// Phase 1: gates[32,2048] = [ctx|emb_t|h] @ [W_ih|W_hh]^T, 8 waves split K, LDS reduce,
//          cell nonlinearity fused (block owns 16 hidden units = all 4 gate strips).
// Barrier: 32-block device-scope atomic barrier (release/acquire + s_sleep spin).
// Phase 2: Qf[32,1024] = h_new @ [iWq|tWq]^T + bias (256 jobs = 128 tiles x 2 K-halves).

__global__ __launch_bounds__(256) void gates_lstm_qproj_kernel(
    const bhalf* __restrict__ Wcomb, const float* __restrict__ b_ih,
    const float* __restrict__ b_hh, const bhalf* __restrict__ ctx_bf,
    const bhalf* __restrict__ emb_bf, const bhalf* __restrict__ h_in_bf,
    bhalf* __restrict__ h_out_bf, float* __restrict__ h_f, float* __restrict__ c_f,
    const bhalf* __restrict__ Wq, const float* __restrict__ ibq,
    const float* __restrict__ tbq, float* __restrict__ Qf,
    float* __restrict__ out, int* __restrict__ cnt, int t) {
  __shared__ float lds[8][8][16][16];   // wave, tile(strip*2+mtile), m, n  (64KB)
  int lane = threadIdx.x & 31, wave = threadIdx.x >> 5;
  int j0 = blockIdx.x << 4;             // 16 hidden units per block
  v8f acc[8];
#pragma unroll
  for (int i = 0; i < 8; ++i) acc[i] = (v8f){0.f, 0.f, 0.f, 0.f, 0.f, 0.f, 0.f, 0.f};
  int kbeg = wave * 256;
  for (int kk = 0; kk < 256; kk += 32) {
    int k0 = kbeg + kk;
    int strideA;
    const bhalf* a0b;
    if (k0 < 1024)      { strideA = 1024;    a0b = ctx_bf + k0; }
    else if (k0 < 1536) { strideA = TT * EE; a0b = emb_bf + (size_t)t * EE + (k0 - 1024); }
    else                { strideA = EE;      a0b = h_in_bf + (k0 - 1536); }
    v16bf a0 = load_tileA(a0b, strideA, lane);
    v16bf a1 = load_tileA(a0b + (size_t)16 * strideA, strideA, lane);
#pragma unroll
    for (int s = 0; s < 4; ++s) {
      const bhalf* bp = Wcomb + (size_t)(s * 512 + j0) * KC + k0;
      __builtin_prefetch((const void*)(bp + 32), 0, 1);
      v16bf bm = load_tileB(bp, KC, lane);
      acc[s * 2 + 0] = __builtin_amdgcn_wmma_f32_16x16x32_bf16(false, a0, false, bm,
                                                               (short)0, acc[s * 2 + 0],
                                                               false, false);
      acc[s * 2 + 1] = __builtin_amdgcn_wmma_f32_16x16x32_bf16(false, a1, false, bm,
                                                               (short)0, acc[s * 2 + 1],
                                                               false, false);
    }
  }
  int n = lane & 15, hi = lane >> 4;
#pragma unroll
  for (int i = 0; i < 8; ++i)
#pragma unroll
    for (int v = 0; v < 8; ++v) lds[wave][i][v + 8 * hi][n] = acc[i][v];
  __syncthreads();
  for (int p = threadIdx.x; p < 512; p += 256) {
    int m = p >> 4, jj = p & 15;
    int mtile = m >> 4, mr = m & 15;
    int j = j0 + jj;
    float g[4];
#pragma unroll
    for (int s = 0; s < 4; ++s) {
      float v = 0.f;
#pragma unroll
      for (int w = 0; w < 8; ++w) v += lds[w][s * 2 + mtile][mr][jj];
      g[s] = v + b_ih[s * 512 + j] + b_hh[s * 512 + j];
    }
    float si = 1.f / (1.f + __expf(-g[0]));
    float sf = 1.f / (1.f + __expf(-g[1]));
    float tg = tanhf(g[2]);
    float so = 1.f / (1.f + __expf(-g[3]));
    float cn = sf * c_f[m * EE + j] + si * tg;
    float hn = so * tanhf(cn);
    c_f[m * EE + j] = cn;
    h_f[m * EE + j] = hn;
    h_out_bf[m * EE + j] = (bhalf)hn;
    out[((size_t)m * TT + t) * 1536 + j] = hn;   // out[b,t,0:512] = h_new
  }
  // ---- device-scope barrier across the 32 blocks ----
  __syncthreads();
  if (threadIdx.x == 0) {
    __hip_atomic_fetch_add(cnt, 1, __ATOMIC_RELEASE, __HIP_MEMORY_SCOPE_AGENT);
    while (__hip_atomic_load(cnt, __ATOMIC_ACQUIRE, __HIP_MEMORY_SCOPE_AGENT) < 32)
      __builtin_amdgcn_s_sleep(1);
  }
  __syncthreads();
  // ---- phase 2: Q projection; 256 jobs = tile(128) x khalf(2); pair combines in LDS ----
  int job = blockIdx.x * 8 + wave;
  int tile = job >> 1, khalf = job & 1;
  int ntile = tile >> 1, mtile2 = tile & 1;
  v8f qa = {0.f, 0.f, 0.f, 0.f, 0.f, 0.f, 0.f, 0.f};
  const bhalf* A0 = h_out_bf + (size_t)mtile2 * 16 * EE + khalf * 256;
  const bhalf* B0 = Wq + (size_t)ntile * 16 * EE + khalf * 256;
  for (int k0 = 0; k0 < 256; k0 += 32) {
    v16bf a = load_tileA(A0 + k0, EE, lane);
    v16bf b = load_tileB(B0 + k0, EE, lane);
    qa = __builtin_amdgcn_wmma_f32_16x16x32_bf16(false, a, false, b, (short)0, qa,
                                                 false, false);
  }
#pragma unroll
  for (int v = 0; v < 8; ++v) lds[wave][0][v + 8 * hi][n] = qa[v];
  __syncthreads();
  if ((wave & 1) == 0) {
    int col = ntile * 16 + n;
    float bias = (col < 512) ? ibq[col] : tbq[col - 512];
#pragma unroll
    for (int v = 0; v < 8; ++v) {
      float val = lds[wave][0][v + 8 * hi][n] + lds[wave + 1][0][v + 8 * hi][n] + bias;
      Qf[(size_t)(mtile2 * 16 + v + 8 * hi) * 1024 + col] = val;
    }
  }
}

// standalone Q projection (used once for ctx0 from h0)
__global__ __launch_bounds__(128) void qproj_kernel(const bhalf* __restrict__ h_bf,
                                                    const bhalf* __restrict__ Wq,
                                                    const float* __restrict__ ibq,
                                                    const float* __restrict__ tbq,
                                                    float* __restrict__ Qf) {
  int lane = threadIdx.x & 31, wave = threadIdx.x >> 5;
  int id = blockIdx.x * 4 + wave;   // 128 tiles: 64 ntiles x 2 mtiles
  int ntile = id >> 1, mtile = id & 1;
  v8f acc = {0.f, 0.f, 0.f, 0.f, 0.f, 0.f, 0.f, 0.f};
  const bhalf* A0 = h_bf + (size_t)mtile * 16 * EE;
  const bhalf* B0 = Wq + (size_t)ntile * 16 * EE;
  for (int k0 = 0; k0 < EE; k0 += 32) {
    v16bf a = load_tileA(A0 + k0, EE, lane);
    v16bf b = load_tileB(B0 + k0, EE, lane);
    acc = __builtin_amdgcn_wmma_f32_16x16x32_bf16(false, a, false, b, (short)0, acc,
                                                  false, false);
  }
  int n = lane & 15, hi = lane >> 4, col = ntile * 16 + n;
  float bias = (col < 512) ? ibq[col] : tbq[col - 512];
#pragma unroll
  for (int v = 0; v < 8; ++v)
    Qf[(size_t)(mtile * 16 + v + 8 * hi) * 1024 + col] = acc[v] + bias;
}

// ---------------- per-step: softmax attention core ----------------
// 512 blocks x 256 threads: att(2) x b(32) x head(8). K rows contiguous; V transposed.
__global__ __launch_bounds__(256) void attn_kernel(
    const float* __restrict__ Qf, const bhalf* __restrict__ Ki, const bhalf* __restrict__ ViT,
    const bhalf* __restrict__ Kt, const bhalf* __restrict__ VtT,
    const unsigned char* __restrict__ maskI, const unsigned char* __restrict__ maskT,
    bhalf* __restrict__ attnout) {
  __shared__ float q[HD];
  __shared__ float sc[SS];
  __shared__ float red[256];
  int bid = blockIdx.x;
  int att = bid >> 8, rem = bid & 255;
  int b = rem >> 3, head = rem & 7;
  const bhalf* K = att ? Kt : Ki;
  const bhalf* VT = att ? VtT : ViT;
  const unsigned char* mk = att ? maskT : maskI;
  int tid = threadIdx.x;
  if (tid < HD) q[tid] = Qf[(size_t)b * 1024 + att * 512 + head * HD + tid];
  __syncthreads();
  const bhalf* Kb = K + ((size_t)b * SS) * EE + head * HD;
  float myS[2], lmax = -1e30f;
#pragma unroll
  for (int i = 0; i < 2; ++i) {
    int k = tid + i * 256;
    const bhalf* kr = Kb + (size_t)k * EE;
    float s = 0.f;
#pragma unroll
    for (int d8 = 0; d8 < HD; d8 += 8) {
      U8 kv8; kv8.u = *(const uint4*)(kr + d8);
#pragma unroll
      for (int j = 0; j < 8; ++j) s += q[d8 + j] * (float)kv8.h[j];
    }
    s *= 0.125f;                       // 1/sqrt(64)
    if (mk[(size_t)b * SS + k]) s = -1e9f;
    myS[i] = s;
    lmax = fmaxf(lmax, s);
  }
  red[tid] = lmax; __syncthreads();
  for (int off = 128; off > 0; off >>= 1) {
    if (tid < off) red[tid] = fmaxf(red[tid], red[tid + off]);
    __syncthreads();
  }
  float mx = red[0]; __syncthreads();
  float lsum = 0.f;
#pragma unroll
  for (int i = 0; i < 2; ++i) {
    float e = __expf(myS[i] - mx);
    sc[tid + i * 256] = e;
    lsum += e;
  }
  red[tid] = lsum; __syncthreads();
  for (int off = 128; off > 0; off >>= 1) {
    if (tid < off) red[tid] += red[tid + off];
    __syncthreads();
  }
  float inv = 1.f / red[0];
  __syncthreads();
  // AV: 4 threads per output dim, contiguous reads from transposed V
  int d = tid >> 2, part = tid & 3;
  const bhalf* Vr = VT + (((size_t)(b * NH + head) * HD + d) * SS) + part * 128;
  float a = 0.f;
#pragma unroll
  for (int k8 = 0; k8 < 128; k8 += 8) {
    U8 v8_; v8_.u = *(const uint4*)(Vr + k8);
#pragma unroll
    for (int j = 0; j < 8; ++j) a += sc[part * 128 + k8 + j] * (float)v8_.h[j];
  }
  red[tid] = a; __syncthreads();
  if (part == 0) {
    float sres = (red[tid] + red[tid + 1] + red[tid + 2] + red[tid + 3]) * inv;
    attnout[(size_t)b * 1024 + att * 512 + head * HD + d] = (bhalf)sres;
  }
}

// ---------------- per-step: output projection + residual; writes ctx (bf16) & out ----------------
__global__ __launch_bounds__(256) void outproj_kernel(
    const bhalf* __restrict__ attnout, const bhalf* __restrict__ Wo,
    const float* __restrict__ ibo, const float* __restrict__ tbo,
    const float* __restrict__ h_f, bhalf* __restrict__ ctx_bf, float* __restrict__ out,
    int t) {
  __shared__ float lds[8][16][16];
  int lane = threadIdx.x & 31, wave = threadIdx.x >> 5;
  int job = blockIdx.x * 8 + wave;  // 256 jobs = 128 tiles x 2 K-halves
  int tile = job >> 1, khalf = job & 1;
  int ntile = tile >> 1, mtile = tile & 1;
  int col0 = ntile * 16, att = col0 >> 9;
  v8f acc = {0.f, 0.f, 0.f, 0.f, 0.f, 0.f, 0.f, 0.f};
  const bhalf* A0 = attnout + (size_t)mtile * 16 * 1024 + att * 512 + khalf * 256;
  const bhalf* B0 = Wo + (size_t)col0 * EE + khalf * 256;
  for (int k0 = 0; k0 < 256; k0 += 32) {
    v16bf a = load_tileA(A0 + k0, 1024, lane);
    v16bf b = load_tileB(B0 + k0, EE, lane);
    acc = __builtin_amdgcn_wmma_f32_16x16x32_bf16(false, a, false, b, (short)0, acc,
                                                  false, false);
  }
  int n = lane & 15, hi = lane >> 4;
#pragma unroll
  for (int v = 0; v < 8; ++v) lds[wave][v + 8 * hi][n] = acc[v];
  __syncthreads();
  if ((wave & 1) == 0) {
    int col = col0 + n;
    float bias = (col < 512) ? ibo[col] : tbo[col - 512];
#pragma unroll
    for (int v = 0; v < 8; ++v) {
      int m = mtile * 16 + v + 8 * hi;
      float val = lds[wave][v + 8 * hi][n] + lds[wave + 1][v + 8 * hi][n] + bias +
                  h_f[(size_t)m * EE + (col & 511)];
      ctx_bf[(size_t)m * 1024 + col] = (bhalf)val;
      if (t >= 0) out[((size_t)m * TT + t) * 1536 + 512 + col] = val;
    }
  }
}

// ---------------- host orchestration ----------------

extern "C" void kernel_launch(void* const* d_in, const int* in_sizes, int n_in,
                              void* d_out, int out_size, void* d_ws, size_t ws_size,
                              hipStream_t stream) {
  (void)in_sizes; (void)n_in; (void)out_size; (void)ws_size;
  const float* emb  = (const float*)d_in[0];
  const float* h0   = (const float*)d_in[1];
  const float* c0   = (const float*)d_in[2];
  const float* encI = (const float*)d_in[3];
  const float* encT = (const float*)d_in[4];
  const unsigned char* maskI = (const unsigned char*)d_in[5];
  const unsigned char* maskT = (const unsigned char*)d_in[6];
  const float* W_ih = (const float*)d_in[7];
  const float* W_hh = (const float*)d_in[8];
  const float* b_ih = (const float*)d_in[9];
  const float* b_hh = (const float*)d_in[10];
  const float* iWq = (const float*)d_in[11]; const float* ibq = (const float*)d_in[12];
  const float* iWk = (const float*)d_in[13]; const float* ibk = (const float*)d_in[14];
  const float* iWv = (const float*)d_in[15]; const float* ibv = (const float*)d_in[16];
  const float* iWo = (const float*)d_in[17]; const float* ibo = (const float*)d_in[18];
  const float* tWq = (const float*)d_in[19]; const float* tbq = (const float*)d_in[20];
  const float* tWk = (const float*)d_in[21]; const float* tbk = (const float*)d_in[22];
  const float* tWv = (const float*)d_in[23]; const float* tbv = (const float*)d_in[24];
  const float* tWo = (const float*)d_in[25]; const float* tbo = (const float*)d_in[26];
  float* out = (float*)d_out;

  char* ws = (char*)d_ws;
  auto alloc = [&](size_t bytes) -> char* {
    char* p = ws;
    ws += (bytes + 255) & ~(size_t)255;
    return p;
  };
  bhalf* Wcomb = (bhalf*)alloc((size_t)G4 * KC * 2);          // 8 MB
  bhalf* Wqc   = (bhalf*)alloc((size_t)1024 * EE * 2);
  bhalf* Woc   = (bhalf*)alloc((size_t)1024 * EE * 2);
  bhalf* embb  = (bhalf*)alloc((size_t)BQ * TT * EE * 2);     // 8 MB
  bhalf* encIb = (bhalf*)alloc((size_t)BQ * SS * EE * 2);     // 16.8 MB
  bhalf* encTb = (bhalf*)alloc((size_t)BQ * SS * EE * 2);
  bhalf* Wki   = (bhalf*)alloc((size_t)EE * EE * 2);
  bhalf* Wvi   = (bhalf*)alloc((size_t)EE * EE * 2);
  bhalf* Wkt   = (bhalf*)alloc((size_t)EE * EE * 2);
  bhalf* Wvt   = (bhalf*)alloc((size_t)EE * EE * 2);
  bhalf* Ki    = (bhalf*)alloc((size_t)BQ * SS * EE * 2);     // 16.8 MB each
  bhalf* ViT   = (bhalf*)alloc((size_t)BQ * SS * EE * 2);
  bhalf* Kt    = (bhalf*)alloc((size_t)BQ * SS * EE * 2);
  bhalf* VtT   = (bhalf*)alloc((size_t)BQ * SS * EE * 2);
  bhalf* ctxb  = (bhalf*)alloc((size_t)BQ * 1024 * 2);
  bhalf* hbA   = (bhalf*)alloc((size_t)BQ * EE * 2);
  bhalf* hbB   = (bhalf*)alloc((size_t)BQ * EE * 2);
  float* hf    = (float*)alloc((size_t)BQ * EE * 4);
  float* cf    = (float*)alloc((size_t)BQ * EE * 4);
  float* Qf    = (float*)alloc((size_t)BQ * 1024 * 4);
  bhalf* atno  = (bhalf*)alloc((size_t)BQ * 1024 * 2);
  int*   cnt   = (int*)alloc(TT * sizeof(int));

  // one-time prep: bf16 weights/activations + KV caches (hoisted out of the time loop)
  zero_kernel<<<1, 256, 0, stream>>>(cnt, TT);
  build_wcomb_kernel<<<4096, 256, 0, stream>>>(W_ih, W_hh, Wcomb);
  build_pair_kernel<<<1024, 256, 0, stream>>>(iWq, tWq, Wqc, EE * EE);
  build_pair_kernel<<<1024, 256, 0, stream>>>(iWo, tWo, Woc, EE * EE);
  cvt_kernel<<<4096, 256, 0, stream>>>(emb, embb, (size_t)BQ * TT * EE);
  cvt_kernel<<<8192, 256, 0, stream>>>(encI, encIb, (size_t)BQ * SS * EE);
  cvt_kernel<<<8192, 256, 0, stream>>>(encT, encTb, (size_t)BQ * SS * EE);
  cvt_kernel<<<512, 256, 0, stream>>>(iWk, Wki, (size_t)EE * EE);
  cvt_kernel<<<512, 256, 0, stream>>>(iWv, Wvi, (size_t)EE * EE);
  cvt_kernel<<<512, 256, 0, stream>>>(tWk, Wkt, (size_t)EE * EE);
  cvt_kernel<<<512, 256, 0, stream>>>(tWv, Wvt, (size_t)EE * EE);
  kv_kernel  <<<8192, 128, 0, stream>>>(encIb, Wki, ibk, Ki);
  kv_kernel_T<<<8192, 128, 0, stream>>>(encIb, Wvi, ibv, ViT);
  kv_kernel  <<<8192, 128, 0, stream>>>(encTb, Wkt, tbk, Kt);
  kv_kernel_T<<<8192, 128, 0, stream>>>(encTb, Wvt, tbv, VtT);
  init_kernel<<<64, 256, 0, stream>>>(h0, c0, hf, cf, hbA);

  // ctx0 = context_of(h0)
  qproj_kernel<<<32, 128, 0, stream>>>(hbA, Wqc, ibq, tbq, Qf);
  attn_kernel<<<512, 256, 0, stream>>>(Qf, Ki, ViT, Kt, VtT, maskI, maskT, atno);
  outproj_kernel<<<32, 256, 0, stream>>>(atno, Woc, ibo, tbo, hf, ctxb, out, -1);

  for (int t = 0; t < TT; ++t) {
    bhalf* hin  = (t & 1) ? hbB : hbA;
    bhalf* hout = (t & 1) ? hbA : hbB;
    gates_lstm_qproj_kernel<<<32, 256, 0, stream>>>(Wcomb, b_ih, b_hh, ctxb, embb, hin,
                                                    hout, hf, cf, Wqc, ibq, tbq, Qf, out,
                                                    cnt + t, t);
    attn_kernel<<<512, 256, 0, stream>>>(Qf, Ki, ViT, Kt, VtT, maskI, maskT, atno);
    outproj_kernel<<<32, 256, 0, stream>>>(atno, Woc, ibo, tbo, hf, ctxb, out, t);
  }
  tail_kernel<<<64, 256, 0, stream>>>(hf, cf, out);
}
